// MomentumQueue_88553635709439
// MI455X (gfx1250) — matrix operand order repaired and checked by reading, not compiled, not run on hardware
//
#include <hip/hip_runtime.h>
#include <math.h>

// ---------------------------------------------------------------------------
// Problem constants (match reference)
// ---------------------------------------------------------------------------
#define NB      1024      // batch rows
#define ND      256       // feature dim
#define NQ      65536     // memory bank rows
#define NK      200       // top-k
#define NCLS    10
#define QCHUNKS 16        // grid split along Q
#define QCHUNK  (NQ / QCHUNKS)     // 4096
#define NTILE   128                // columns (memory rows) per inner tile
#define NTITER  (QCHUNK / NTILE)   // 32
#define MTILE   32                 // x rows per workgroup

typedef __attribute__((ext_vector_type(16))) __bf16        v16bf;
typedef __attribute__((ext_vector_type(8)))  float         v8f;
typedef __attribute__((ext_vector_type(4)))  unsigned int  v4u;
typedef __attribute__((ext_vector_type(2)))  unsigned int  v2u;

union Frag {
    v4u   q[2];   // two 16B LDS loads -> ds_load_b128
    v16bf v;      // WMMA operand view
};

// f32 -> bf16 round-to-nearest-even (bit trick)
__device__ __forceinline__ unsigned short f2bf(float x) {
    unsigned int u = __float_as_uint(x);
    unsigned int r = u + 0x7FFFu + ((u >> 16) & 1u);
    return (unsigned short)(r >> 16);
}
__device__ __forceinline__ unsigned int pk_bf2(float a, float b) {
    return (unsigned int)f2bf(a) | ((unsigned int)f2bf(b) << 16);
}

// Min-heap (root = smallest kept value) of size NK over (value, index).
// Caller guarantees v > hv[0].
__device__ __forceinline__ void heap_replace_root(float* hv, int* hi, float v, int idx) {
    hv[0] = v; hi[0] = idx;
    int i = 0;
    for (;;) {
        int l = 2 * i + 1;
        if (l >= NK) break;
        int s = l;
        int r = l + 1;
        if (r < NK && hv[r] < hv[l]) s = r;
        if (hv[s] >= hv[i]) break;
        float tv = hv[i]; hv[i] = hv[s]; hv[s] = tv;
        int   ti = hi[i]; hi[i] = hi[s]; hi[s] = ti;
        i = s;
    }
}

// ---------------------------------------------------------------------------
// Kernel 1: L2-normalize rows of x, pack to bf16 (2 per uint) into ws.
// One wave per row; 8 rows per 256-thread block.
// ---------------------------------------------------------------------------
__global__ void nce_normalize_bf16(const float* __restrict__ x,
                                   unsigned int* __restrict__ xn) {
    const int wid  = threadIdx.x >> 5;
    const int lane = threadIdx.x & 31;
    const int row  = blockIdx.x * 8 + wid;

    const float4* xr = (const float4*)(x + (size_t)row * ND);
    float4 f0 = xr[lane * 2 + 0];
    float4 f1 = xr[lane * 2 + 1];

    float s = f0.x * f0.x + f0.y * f0.y + f0.z * f0.z + f0.w * f0.w +
              f1.x * f1.x + f1.y * f1.y + f1.z * f1.z + f1.w * f1.w;
    for (int o = 16; o > 0; o >>= 1) s += __shfl_xor(s, o);
    const float sc = 1.0f / fmaxf(sqrtf(s), 1e-12f);

    unsigned int* dst = xn + (size_t)row * (ND / 2) + lane * 4;
    dst[0] = pk_bf2(f0.x * sc, f0.y * sc);
    dst[1] = pk_bf2(f0.z * sc, f0.w * sc);
    dst[2] = pk_bf2(f1.x * sc, f1.y * sc);
    dst[3] = pk_bf2(f1.z * sc, f1.w * sc);
}

// ---------------------------------------------------------------------------
// Kernel 2: fused bf16-WMMA GEMM tile + per-row top-200 selection.
// grid = (NB/MTILE, QCHUNKS), block = 256 threads (8 waves).
// Dynamic LDS layout (bytes):
//   As    @ 0      : MTILE x ND bf16          = 16384
//   Bs    @ 16384  : NTILE x ND bf16          = 65536
//   stage @ 81920  : MTILE x NTILE f32        = 16384
//   heapV @ 98304  : MTILE x NK f32           = 25600
//   heapI @ 123904 : MTILE x NK i32           = 25600
//   total = 149504
// ---------------------------------------------------------------------------
#define KNN_LDS_BYTES 149504

__global__ void __launch_bounds__(256)
nce_knn_topk(const float* __restrict__ mem,
             const unsigned int* __restrict__ xn,
             float* __restrict__ candV,
             int*   __restrict__ candI) {
    extern __shared__ char smem[];
    unsigned int* AsU   = (unsigned int*)(smem);              // [MTILE][ND/2]
    unsigned int* BsU   = (unsigned int*)(smem + 16384);      // [NTILE][ND/2]
    float*        stage = (float*)(smem + 81920);             // [MTILE][NTILE]
    float*        heapV = (float*)(smem + 98304);             // [MTILE][NK]
    int*          heapI = (int*)  (smem + 123904);            // [MTILE][NK]

    const int tid   = threadIdx.x;
    const int lane  = tid & 31;
    const int wid   = tid >> 5;
    const int r0    = blockIdx.x * MTILE;
    const int qBase = blockIdx.y * QCHUNK;

    // ---- Stage normalized-x tile into LDS via async global->LDS DMA -------
    // 16 KB = 1024 x b128 transfers; 4 per thread. ASYNCcnt-tracked, no VGPR
    // round trip (gfx1250 GLOBAL_LOAD_ASYNC_TO_LDS_B128).
    {
        const unsigned int* src = xn + (size_t)r0 * (ND / 2);
        const unsigned int  lds_base = (unsigned int)(size_t)(void*)AsU; // LDS offset in addr[31:0]
#pragma unroll
        for (int i = 0; i < 4; ++i) {
            int e = tid + i * 256;                       // b128 index
            unsigned int       loff = lds_base + (unsigned int)(e * 16);
            unsigned long long gptr = (unsigned long long)(size_t)(src + e * 4);
            asm volatile("global_load_async_to_lds_b128 %0, %1, off"
                         :: "v"(loff), "v"(gptr) : "memory");
        }
    }
    // Init heaps to -inf sentinel while the DMA is in flight.
    for (int e = tid; e < MTILE * NK; e += 256) { heapV[e] = -1e30f; heapI[e] = 0; }
    asm volatile("s_wait_asynccnt 0x0" ::: "memory");
    __syncthreads();

    const int hs = lane >> 4;      // half-select (lanes 16..31)
    const int rr = lane & 15;

    for (int nt = 0; nt < NTITER; ++nt) {
        const int g0 = qBase + nt * NTILE;

        // --- load 128x256 f32 memory tile, convert to bf16 in LDS -----------
        const float4* m4 = (const float4*)mem + (size_t)g0 * (ND / 4);
        v2u* Bs2 = (v2u*)BsU;
        for (int i = 0; i < 32; ++i) {
            int e = tid + i * 256;            // 0..8191 float4s
            float4 f = m4[e];
            v2u p; p.x = pk_bf2(f.x, f.y); p.y = pk_bf2(f.z, f.w);
            Bs2[e] = p;                       // ds_store_b64
        }
        if (nt + 1 < NTITER)                  // gfx1250 global_prefetch_b8
            __builtin_prefetch((const void*)((const float4*)mem +
                               (size_t)(g0 + NTILE) * (ND / 4) + tid), 0, 0);
        __syncthreads();

        // --- 32x128 score tile: each wave owns one 16-col subtile,
        //     two 16-row subtiles (B fragment reused). -----------------------
        {
            v8f c0 = {};
            v8f c1 = {};
            const int bn = (wid << 4) + rr;   // memory row within tile (B col)
            const v4u* Aq = (const v4u*)AsU;  // [MTILE][32] b128 view
            const v4u* Bq = (const v4u*)BsU;  // [NTILE][32] b128 view
#pragma unroll
            for (int k0 = 0; k0 < ND; k0 += 32) {
                Frag b;                       // two 4-uint runs / lane
                const v4u* bp = Bq + bn * 32 + (k0 >> 3) + hs;
                b.q[0] = bp[0];
                b.q[1] = bp[2];

                Frag a;                       // two 4-uint runs / lane
                const v4u* ap0 = Aq + rr * 32 + (k0 >> 3) + hs;
                a.q[0] = ap0[0];
                a.q[1] = ap0[2];
                c0 = __builtin_amdgcn_wmma_f32_16x16x32_bf16(
                        false, a.v, false, b.v, (short)0, c0, false, false);

                const v4u* ap1 = Aq + (16 + rr) * 32 + (k0 >> 3) + hs;
                a.q[0] = ap1[0];
                a.q[1] = ap1[2];
                c1 = __builtin_amdgcn_wmma_f32_16x16x32_bf16(
                        false, a.v, false, b.v, (short)0, c1, false, false);
            }
            const int colBase = (wid << 4) + rr;
#pragma unroll
            for (int v = 0; v < 8; ++v) {
                stage[(v + (hs << 3)) * NTILE + colBase]      = c0[v];
                stage[(16 + v + (hs << 3)) * NTILE + colBase] = c1[v];
            }
        }
        __syncthreads();

        // --- per-row top-K maintenance (wave 0: one thread per row) ---------
        if (tid < MTILE) {
            float* hv = heapV + tid * NK;
            int*   hi = heapI + tid * NK;
            const float* srow = stage + tid * NTILE;
            for (int j = 0; j < NTILE; ++j) {
                float v = srow[j];
                if (v > hv[0]) heap_replace_root(hv, hi, v, g0 + j);
            }
        }
        __syncthreads();
    }

    // Write per-(row, chunk) top-200 candidates to workspace.
    for (int e = tid; e < MTILE * NK; e += 256) {
        int r = e / NK, j = e % NK;
        size_t o = ((size_t)(r0 + r) * QCHUNKS + blockIdx.y) * NK + j;
        candV[o] = heapV[e];
        candI[o] = heapI[e];
    }
}

// ---------------------------------------------------------------------------
// Kernel 3: merge 16x200 candidates per row -> global top-200 -> softmax/0.07
// -> label-weighted class scores. One thread per row; 32 rows per block.
// ---------------------------------------------------------------------------
__global__ void nce_merge_softmax(const float* __restrict__ candV,
                                  const int*   __restrict__ candI,
                                  const int*   __restrict__ labels,
                                  float*       __restrict__ out) {
    __shared__ float hV[32 * NK];
    __shared__ int   hI[32 * NK];
    const int t   = threadIdx.x;
    const int row = blockIdx.x * 32 + t;

    float* hv = hV + t * NK;
    int*   hi = hI + t * NK;
    for (int j = 0; j < NK; ++j) { hv[j] = -1e30f; hi[j] = 0; }

    const float* cv = candV + (size_t)row * (QCHUNKS * NK);
    const int*   ci = candI + (size_t)row * (QCHUNKS * NK);
    for (int j = 0; j < QCHUNKS * NK; ++j) {
        float v = cv[j];
        if (v > hv[0]) heap_replace_root(hv, hi, v, ci[j]);
    }

    float m = -1e30f;
    for (int j = 0; j < NK; ++j) m = fmaxf(m, hv[j]);

    float cls[NCLS];
#pragma unroll
    for (int c = 0; c < NCLS; ++c) cls[c] = 0.0f;

    const float invT = 1.0f / 0.07f;
    float sum = 0.0f;
    for (int j = 0; j < NK; ++j) {
        float w = __expf((hv[j] - m) * invT);
        sum += w;
        int lb = labels[hi[j]];
#pragma unroll
        for (int c = 0; c < NCLS; ++c) cls[c] += (lb == c) ? w : 0.0f;
    }
    const float inv = 1.0f / sum;
#pragma unroll
    for (int c = 0; c < NCLS; ++c)
        out[(size_t)row * NCLS + c] = fminf(cls[c] * inv + 1e-5f, 1.0f);
}

// ---------------------------------------------------------------------------
// Launch
// ---------------------------------------------------------------------------
extern "C" void kernel_launch(void* const* d_in, const int* in_sizes, int n_in,
                              void* d_out, int out_size, void* d_ws, size_t ws_size,
                              hipStream_t stream) {
    const float* x      = (const float*)d_in[0];
    const float* mem    = (const float*)d_in[1];
    const int*   labels = (const int*)d_in[2];
    float*       out    = (float*)d_out;

    char* ws = (char*)d_ws;
    // ws layout: xn bf16 (512 KB) | candV (13.1 MB) | candI (13.1 MB)
    unsigned int* xn    = (unsigned int*)ws;
    const size_t XN_BYTES   = (size_t)NB * (ND / 2) * sizeof(unsigned int); // 524288
    const size_t CAND_ELEMS = (size_t)NB * QCHUNKS * NK;                    // 3,276,800
    float* candV = (float*)(ws + XN_BYTES);
    int*   candI = (int*)  (ws + XN_BYTES + CAND_ELEMS * sizeof(float));

    nce_normalize_bf16<<<NB / 8, 256, 0, stream>>>(x, xn);

    dim3 grid(NB / MTILE, QCHUNKS);
    nce_knn_topk<<<grid, 256, KNN_LDS_BYTES, stream>>>(mem, xn, candV, candI);

    nce_merge_softmax<<<NB / 32, 32, 0, stream>>>(candV, candI, labels, out);
}
